// LookupTablePosMy_49641232007203
// MI455X (gfx1250) — matrix operand order repaired and checked by reading, not compiled, not run on hardware
//
#include <hip/hip_runtime.h>
#include <cfloat>
#include <cmath>

// ---------------------------------------------------------------------------
// Problem constants (from reference)
// ---------------------------------------------------------------------------
#define BN   4096      // batch
#define EV   128       // embedding dim
#define SEQ  19        // 4*K-1
#define NN2  20        // input2 length
#define HD   1920      // LSTM hidden (3*K*E)
#define G4   7680      // 4*H
#define E2   256       // 2*E (LSTM input size)

typedef unsigned short u16;
typedef __attribute__((ext_vector_type(16))) __bf16 v16bf;
typedef __attribute__((ext_vector_type(8)))  __bf16 v8bf;
typedef __attribute__((ext_vector_type(8)))  float  v8f;

// f32 -> bf16 with round-to-nearest-even (bit-level, host-independent)
__device__ __forceinline__ u16 f2bf(float x) {
    unsigned u = __builtin_bit_cast(unsigned, x);
    unsigned r = u + 0x7FFFu + ((u >> 16) & 1u);
    return (u16)(r >> 16);
}

// ---------------------------------------------------------------------------
// WMMA fragment loaders (layouts per CDNA5 ISA 7.12.2, wave32)
// A (16x32 bf16, MxK): lane L: m = L%16, half = L/16;
//   vec elems [0..7]  = A[m][ 8*half + 0..7]
//   vec elems [8..15] = A[m][16 + 8*half + 0..7]       (two 16B loads)
// B (32x16 bf16, KxN) from row-major W (NxK): B[k][n] = W[n][k]; lane L:
//   n = L%16, half = L/16; vec elems [0..7] = W[n][8*half + 0..7],
//   [8..15] = W[n][16 + 8*half + 0..7]   (two 16B loads)
// ---------------------------------------------------------------------------
__device__ __forceinline__ v16bf load_a_frag(const u16* A, int lda, int m,
                                             int k0, int half) {
    const u16* p = A + (size_t)m * lda + k0 + half * 8;
    v8bf lo = *reinterpret_cast<const v8bf*>(p);
    v8bf hi = *reinterpret_cast<const v8bf*>(p + 16);
    return __builtin_shufflevector(lo, hi, 0,1,2,3,4,5,6,7,8,9,10,11,12,13,14,15);
}

__device__ __forceinline__ v16bf load_b_frag(const u16* W, int ldw, int n,
                                             int k0, int half) {
    const u16* p = W + (size_t)n * ldw + k0 + half * 8;
    v8bf lo = *reinterpret_cast<const v8bf*>(p);
    v8bf hi = *reinterpret_cast<const v8bf*>(p + 16);
    return __builtin_shufflevector(lo, hi, 0,1,2,3,4,5,6,7,8,9,10,11,12,13,14,15);
}

#define WMMA_BF16(a, b, c) \
    __builtin_amdgcn_wmma_f32_16x16x32_bf16(false, (a), false, (b), (short)0, (c), false, false)

// ---------------------------------------------------------------------------
// Kernel 1: embeddings + positional -> bf16 X, laid out (SEQ, B, 2E)
// ---------------------------------------------------------------------------
__global__ __launch_bounds__(256) void embed_pos_kernel(
        const int* __restrict__ input1, const float* __restrict__ emb,
        const float* __restrict__ pos, u16* __restrict__ Xbf) {
    int row = blockIdx.x;            // row = s*BN + b
    int s = row / BN;
    int b = row % BN;
    int t = threadIdx.x;             // 0..255
    u16* dst = Xbf + (size_t)row * E2;
    if (t < EV) {
        int tok = input1[b * SEQ + s];
        dst[t] = f2bf(emb[(size_t)tok * EV + t]);
    } else {
        dst[t] = f2bf(pos[s * EV + (t - EV)]);
    }
}

// ---------------------------------------------------------------------------
// Kernel 2: f32 -> bf16 weight conversion (grid-stride)
// ---------------------------------------------------------------------------
__global__ __launch_bounds__(256) void f32_to_bf16_kernel(
        const float* __restrict__ src, u16* __restrict__ dst, int n) {
    for (int i = blockIdx.x * blockDim.x + threadIdx.x; i < n;
         i += gridDim.x * blockDim.x)
        dst[i] = f2bf(src[i]);
}

// ---------------------------------------------------------------------------
// Kernel 3: zero fill (16B granules, grid-stride)
// ---------------------------------------------------------------------------
__global__ __launch_bounds__(256) void zero16_kernel(float4* __restrict__ p, int n16) {
    float4 z; z.x = 0.f; z.y = 0.f; z.z = 0.f; z.w = 0.f;
    for (int i = blockIdx.x * blockDim.x + threadIdx.x; i < n16;
         i += gridDim.x * blockDim.x)
        p[i] = z;
}

// ---------------------------------------------------------------------------
// Kernel 4: gates = Xt @ Wih^T + H @ Whh^T   (bf16 WMMA, f32 accumulate)
//   Xt: (BN, 256) bf16, Wih: (7680, 256) bf16 row-major
//   H : (BN,1920) bf16, Whh: (7680,1920) bf16 row-major
//   gates: (BN, 7680) f32
// Block: 256 threads = 8 waves laid out 2(M) x 4(N); wave tile 32x64;
// block tile 64x256; grid 64 x 30 = 1920 blocks.
// ---------------------------------------------------------------------------
__device__ __forceinline__ void gemm_accum(const u16* __restrict__ A, int lda,
                                           const u16* __restrict__ W, int K,
                                           int mBase, int nBase,
                                           int nl, int half, v8f acc[2][4]) {
    for (int k0 = 0; k0 < K; k0 += 32) {
        v16bf a0 = load_a_frag(A, lda, mBase + nl,      k0, half);
        v16bf a1 = load_a_frag(A, lda, mBase + 16 + nl, k0, half);
        v16bf b0 = load_b_frag(W, lda, nBase + 0  + nl, k0, half);
        v16bf b1 = load_b_frag(W, lda, nBase + 16 + nl, k0, half);
        v16bf b2 = load_b_frag(W, lda, nBase + 32 + nl, k0, half);
        v16bf b3 = load_b_frag(W, lda, nBase + 48 + nl, k0, half);
        acc[0][0] = WMMA_BF16(a0, b0, acc[0][0]);
        acc[0][1] = WMMA_BF16(a0, b1, acc[0][1]);
        acc[0][2] = WMMA_BF16(a0, b2, acc[0][2]);
        acc[0][3] = WMMA_BF16(a0, b3, acc[0][3]);
        acc[1][0] = WMMA_BF16(a1, b0, acc[1][0]);
        acc[1][1] = WMMA_BF16(a1, b1, acc[1][1]);
        acc[1][2] = WMMA_BF16(a1, b2, acc[1][2]);
        acc[1][3] = WMMA_BF16(a1, b3, acc[1][3]);
    }
}

__global__ __launch_bounds__(256) void lstm_gemm_kernel(
        const u16* __restrict__ Xt,  const u16* __restrict__ Wih,
        const u16* __restrict__ Hb,  const u16* __restrict__ Whh,
        float* __restrict__ gates) {
    const int lane = threadIdx.x & 31;
    const int nl   = lane & 15;
    const int half = lane >> 4;
    const int w    = threadIdx.x >> 5;
    const int mBase = (blockIdx.x % 64) * 64 + (w >> 2) * 32;   // M tile
    const int nBase = (blockIdx.x / 64) * 256 + (w & 3) * 64;   // N tile

    v8f acc[2][4];
    #pragma unroll
    for (int i = 0; i < 2; ++i)
        #pragma unroll
        for (int j = 0; j < 4; ++j)
            acc[i][j] = (v8f){0.f,0.f,0.f,0.f,0.f,0.f,0.f,0.f};

    gemm_accum(Xt, E2, Wih, E2, mBase, nBase, nl, half, acc);   // x part, K=256
    gemm_accum(Hb, HD, Whh, HD, mBase, nBase, nl, half, acc);   // h part, K=1920

    // D layout (32-bit C/D 16x16): VGPR r -> row = r + 8*half, col = lane%16
    #pragma unroll
    for (int i = 0; i < 2; ++i) {
        #pragma unroll
        for (int j = 0; j < 4; ++j) {
            float* p = gates + (size_t)(mBase + i * 16 + half * 8) * G4
                             + (nBase + j * 16 + nl);
            #pragma unroll
            for (int r = 0; r < 8; ++r)
                p[(size_t)r * G4] = acc[i][j][r];
        }
    }
}

// ---------------------------------------------------------------------------
// Kernel 5: LSTM cell elementwise (i,f,g,o split along 4H as in reference)
// ---------------------------------------------------------------------------
__global__ __launch_bounds__(256) void lstm_cell_kernel(
        const float* __restrict__ gates, const float* __restrict__ bih,
        const float* __restrict__ bhh, float* __restrict__ h,
        float* __restrict__ c, u16* __restrict__ Hb) {
    int idx = blockIdx.x * blockDim.x + threadIdx.x;
    if (idx >= BN * HD) return;
    int b = idx / HD;
    int j = idx % HD;
    const float* g = gates + (size_t)b * G4;
    float gi = g[j]          + bih[j]          + bhh[j];
    float gf = g[j + HD]     + bih[j + HD]     + bhh[j + HD];
    float gg = g[j + 2 * HD] + bih[j + 2 * HD] + bhh[j + 2 * HD];
    float go = g[j + 3 * HD] + bih[j + 3 * HD] + bhh[j + 3 * HD];
    float i_ = 1.f / (1.f + __expf(-gi));
    float f_ = 1.f / (1.f + __expf(-gf));
    float g_ = tanhf(gg);
    float o_ = 1.f / (1.f + __expf(-go));
    float cn = f_ * c[idx] + i_ * g_;
    float hn = o_ * tanhf(cn);
    c[idx] = cn;
    h[idx] = hn;
    Hb[idx] = f2bf(hn);
}

// ---------------------------------------------------------------------------
// Kernel 6: final scoring. One block per batch row.
//   ts = h.reshape(5,384); emb2[i] = concat(emb[input2[b,i+j]], j=0..2)
//   rs[k,i] = dot(ts[k], emb2[i]);  ms = max;  logits = log_softmax(ms*w + b)
// ---------------------------------------------------------------------------
__global__ __launch_bounds__(128) void final_kernel(
        const float* __restrict__ h, const int* __restrict__ input2,
        const float* __restrict__ emb, const float* __restrict__ lin_w,
        const float* __restrict__ lin_b, float* __restrict__ out) {
    int b = blockIdx.x;
    int t = threadIdx.x;
    __shared__ float red[128];
    float val = -FLT_MAX;
    if (t < 90) {
        int k = t / 18;
        int i = t % 18;
        const float* ts = h + (size_t)b * HD + k * 384;
        float s = 0.f;
        #pragma unroll
        for (int j = 0; j < 3; ++j) {
            int tok = input2[b * NN2 + i + j];
            const float* e  = emb + (size_t)tok * EV;
            const float* tp = ts + j * EV;
            #pragma unroll 4
            for (int d = 0; d < EV; ++d) s += tp[d] * e[d];
        }
        val = s;
    }
    red[t] = val;
    __syncthreads();
    for (int s2 = 64; s2 > 0; s2 >>= 1) {
        if (t < s2) red[t] = fmaxf(red[t], red[t + s2]);
        __syncthreads();
    }
    if (t == 0) {
        float ms = red[0];
        float z0 = ms * lin_w[0] + lin_b[0];
        float z1 = ms * lin_w[1] + lin_b[1];
        float m  = fmaxf(z0, z1);
        float lse = m + logf(__expf(z0 - m) + __expf(z1 - m));
        out[2 * b + 0] = z0 - lse;
        out[2 * b + 1] = z1 - lse;
    }
}

// ---------------------------------------------------------------------------
// Host launcher
// ---------------------------------------------------------------------------
extern "C" void kernel_launch(void* const* d_in, const int* in_sizes, int n_in,
                              void* d_out, int out_size, void* d_ws, size_t ws_size,
                              hipStream_t stream) {
    const int*   input1 = (const int*)  d_in[0];
    const int*   input2 = (const int*)  d_in[1];
    const float* emb    = (const float*)d_in[2];
    const float* pos    = (const float*)d_in[3];
    const float* W_ih   = (const float*)d_in[4];
    const float* W_hh   = (const float*)d_in[5];
    const float* b_ih   = (const float*)d_in[6];
    const float* b_hh   = (const float*)d_in[7];
    const float* lin_w  = (const float*)d_in[8];
    const float* lin_b  = (const float*)d_in[9];
    float* out = (float*)d_out;

    // bump allocator over d_ws (all sizes are multiples of 256B)
    char* ws = (char*)d_ws;
    size_t off = 0;
    auto alloc = [&](size_t bytes) -> void* {
        void* p = ws + off;
        off += (bytes + 255) & ~(size_t)255;
        return p;
    };
    u16*   Xbf   = (u16*)  alloc((size_t)SEQ * BN * E2 * 2); //  39.8 MB
    u16*   Wihb  = (u16*)  alloc((size_t)G4 * E2 * 2);       //   3.9 MB
    u16*   Whhb  = (u16*)  alloc((size_t)G4 * HD * 2);       //  29.5 MB
    u16*   Hbf   = (u16*)  alloc((size_t)BN * HD * 2);       //  15.7 MB
    float* hbuf  = (float*)alloc((size_t)BN * HD * 4);       //  31.5 MB
    float* cbuf  = (float*)alloc((size_t)BN * HD * 4);       //  31.5 MB
    float* gates = (float*)alloc((size_t)BN * G4 * 4);       // 125.8 MB

    // setup: embeddings -> bf16 X, weights -> bf16, h/c/Hbf = 0
    embed_pos_kernel<<<BN * SEQ, 256, 0, stream>>>(input1, emb, pos, Xbf);
    f32_to_bf16_kernel<<<2048, 256, 0, stream>>>(W_ih, Wihb, G4 * E2);
    f32_to_bf16_kernel<<<8192, 256, 0, stream>>>(W_hh, Whhb, G4 * HD);
    zero16_kernel<<<4096, 256, 0, stream>>>((float4*)hbuf, BN * HD / 4);
    zero16_kernel<<<4096, 256, 0, stream>>>((float4*)cbuf, BN * HD / 4);
    zero16_kernel<<<4096, 256, 0, stream>>>((float4*)Hbf,  BN * HD / 8);

    // 19 sequential LSTM steps: fused (x,h) GEMM then elementwise cell
    for (int t = 0; t < SEQ; ++t) {
        lstm_gemm_kernel<<<64 * 30, 256, 0, stream>>>(
            Xbf + (size_t)t * BN * E2, Wihb, Hbf, Whhb, gates);
        lstm_cell_kernel<<<(BN * HD) / 256, 256, 0, stream>>>(
            gates, b_ih, b_hh, hbuf, cbuf, Hbf);
    }

    // final scoring + log_softmax
    final_kernel<<<BN, 128, 0, stream>>>(hbuf, input2, emb, lin_w, lin_b, out);
}
